// Experts_24446953848934
// MI455X (gfx1250) — matrix operand order repaired and checked by reading, not compiled, not run on hardware
//
#include <hip/hip_runtime.h>
#include <hip/hip_bf16.h>

// ---- problem constants (match reference) ----
#define E    16
#define C    256
#define CAP  4
#define F    (C * CAP)   // 1024
#define BSZ  8
#define L    1024
#define LT   64          // L tile per workgroup
#define FC   64          // F chunk
#define XPAD 8
#define HPAD 8

typedef __attribute__((ext_vector_type(16))) __bf16       v16bf;
typedef __attribute__((ext_vector_type(8)))  __bf16       v8bf;
typedef __attribute__((ext_vector_type(8)))  float        v8f;
typedef __attribute__((ext_vector_type(4)))  int          v4i;
typedef __attribute__((ext_vector_type(8)))  int          v8i;
typedef __attribute__((ext_vector_type(4)))  unsigned int v4u;

// ---------------- gfx1250 feature probes -------------
#if __has_builtin(__builtin_amdgcn_global_load_async_to_lds_b128)
#define HAVE_ASYNC_LDS 1
#else
#define HAVE_ASYNC_LDS 0
#pragma message("CDNA5 probe: MISSING global_load_async_to_lds_b128")
#endif

#if __has_builtin(__builtin_amdgcn_s_wait_asynccnt)
#define WAIT_ASYNC() __builtin_amdgcn_s_wait_asynccnt(0)
#else
#define WAIT_ASYNC() asm volatile("s_wait_asynccnt 0" ::: "memory")
#endif

#if __has_builtin(__builtin_amdgcn_tensor_load_to_lds) && \
    __has_builtin(__builtin_amdgcn_s_wait_tensorcnt)
#define HAVE_TDM 1
#if __has_include(<hip/amd_detail/amd_gfx1250_TDM.h>)
#define TDM_6ARG 1
#pragma message("CDNA5 probe: TDM 6-arg form (therock headers present)")
#else
#define TDM_6ARG 0
#pragma message("CDNA5 probe: TDM 5-arg form (ROCm, no TDM header)")
#endif
#else
#define HAVE_TDM 0
#pragma message("CDNA5 probe: MISSING tensor_load_to_lds")
#endif

// address-space cast helpers (integer round-trip: generic -> AS1/AS3),
// typed as int4* to match the b128 async builtin's parameter types.
#define AS1P4(p) ((__attribute__((address_space(1))) v4i*)(unsigned long long)(p))
#define AS3P4(p) ((__attribute__((address_space(3))) v4i*)(unsigned long long)(p))

__device__ __forceinline__ v8f wmma_bf16(v16bf a, v16bf b, v8f c) {
  return __builtin_amdgcn_wmma_f32_16x16x32_bf16(
      /*neg_a=*/false, a, /*neg_b=*/false, b,
      /*c_mod=*/(short)0, c, /*reuse_a=*/false, /*reuse_b=*/false);
}

__device__ __forceinline__ v16bf cat8(v8bf lo, v8bf hi) {
  return __builtin_shufflevector(lo, hi, 0, 1, 2, 3, 4, 5, 6, 7,
                                 8, 9, 10, 11, 12, 13, 14, 15);
}

#if HAVE_TDM
// TDM copy of n contiguous f32 (n <= 65535*... here n <= 1024) global -> LDS.
// D# per CDNA5 ISA ch.8: 2-D descriptor, 1 row of n elements, data_size=4B.
__device__ __forceinline__ void tdm_load_row_f32(const float* gsrc,
                                                 void* lds_dst, int n) {
  unsigned long long ga = (unsigned long long)gsrc;
  unsigned int la = (unsigned int)(unsigned long long)lds_dst; // LDS byte addr
  v4u g0;
  g0[0] = 1u;                                    // count=1, user descriptor
  g0[1] = la;                                    // lds_addr
  g0[2] = (unsigned int)(ga & 0xFFFFFFFFu);      // global_addr[31:0]
  g0[3] = (unsigned int)((ga >> 32) & 0x1FFFFFFu) | (2u << 30); // addr hi|type=2
  unsigned int un = (unsigned int)n;
  v8i g1;
  g1[0] = 0x20000;                               // data_size=2 (4 bytes)
  g1[1] = (int)((un & 0xFFFFu) << 16);           // tensor_dim0[15:0]
  g1[2] = (int)((un >> 16) | (1u << 16));        // tensor_dim0 hi | tensor_dim1=1
  g1[3] = (int)((un & 0xFFFFu) << 16);           // tile_dim0 = n
  g1[4] = 1;                                     // tile_dim1 = 1
  g1[5] = (int)un;                               // tensor_dim0_stride = n
  g1[6] = 0;
  g1[7] = 0;
  v4i gz = {0, 0, 0, 0};
#if TDM_6ARG
  v8i gz8 = {0, 0, 0, 0, 0, 0, 0, 0};
  __builtin_amdgcn_tensor_load_to_lds(g0, g1, gz, gz, gz8, 0);
#else
  __builtin_amdgcn_tensor_load_to_lds(g0, g1, gz, gz, 0);
#endif
}
#endif

// ---------------- weight fp32 -> bf16 conversion ----------------
__global__ void cvt_f32_to_bf16(const float* __restrict__ s,
                                __bf16* __restrict__ d, int n) {
  int i = (blockIdx.x * 256 + threadIdx.x) * 4;
  if (i + 3 < n) {
    float4 v = *(const float4*)(s + i);
    unsigned int a0 = __builtin_bit_cast(unsigned short, (__bf16)v.x);
    unsigned int a1 = __builtin_bit_cast(unsigned short, (__bf16)v.y);
    unsigned int a2 = __builtin_bit_cast(unsigned short, (__bf16)v.z);
    unsigned int a3 = __builtin_bit_cast(unsigned short, (__bf16)v.w);
    uint2 o;
    o.x = a0 | (a1 << 16);
    o.y = a2 | (a3 << 16);
    *(uint2*)(d + i) = o;
  }
}

// ---------------- fused grouped-expert MLP ----------------
// grid: BSZ * E * (L/LT) workgroups, 256 threads (8 wave32) each.
__global__ __launch_bounds__(256)
void moe_fused(const float* __restrict__ x,
               const __bf16* __restrict__ w1,
               const __bf16* __restrict__ w2,
               const float* __restrict__ b1,
               const float* __restrict__ b2,
               float* __restrict__ y) {
  __shared__ __bf16 xs[LT][C + XPAD];   // x tile, transposed [l][c]
  __shared__ __bf16 hs[LT][FC + HPAD];  // silu(h) chunk, [l][f]
  __shared__ float  b1s[F];
  __shared__ float  b2s[C];
#if HAVE_ASYNC_LDS
  __shared__ float  xstage[64][64];     // async bounce buffer (16 KB)
#endif

  const int tid   = threadIdx.x;
  const int lane  = tid & 31;
  const int wave  = tid >> 5;       // 0..7
  const int lhalf = lane >> 4;      // 0|1: which K-half this lane owns
  const int lmod  = lane & 15;      // row (A) / column (B,C,D) within tile

  const int nLT = L / LT;           // 16
  const int wg  = blockIdx.x;
  const int lt  = wg % nLT;
  const int e   = (wg / nLT) % E;
  const int b   = wg / (nLT * E);
  const int l0  = lt * LT;

  const float*  xb  = x  + (size_t)(b * E + e) * C * L;
  const __bf16* w1e = w1 + (size_t)e * F * C;     // [F][C] row-major
  const __bf16* w2e = w2 + (size_t)e * C * F;     // [C][F] row-major
  float*        yb  = y  + (size_t)(b * E + e) * C * L;

  // ---- stage biases (TDM DMA if available; visibility via first barrier) ----
#if HAVE_TDM
  if (wave == 0) {
    tdm_load_row_f32(b1 + e * F, &b1s[0], F);
    tdm_load_row_f32(b2 + e * C, &b2s[0], C);
    __builtin_amdgcn_s_wait_tensorcnt(0);
  }
#else
  for (int i = tid; i < F; i += 256) b1s[i] = b1[e * F + i];
  if (tid < C) b2s[tid] = b2[e * C + tid];
#endif

  // ---- stage x tile (fp32 -> bf16, transposed to [l][c]) ----
#if HAVE_ASYNC_LDS
  // ASYNCcnt-tracked copies global->LDS (no VGPR data path), then convert.
  for (int c0s = 0; c0s < C; c0s += 64) {
#pragma unroll
    for (int i = 0; i < 4; ++i) {
      int lin = tid + i * 256;          // 0..1023 float4 slots
      int cc  = lin >> 4;               // 0..63
      int ll  = (lin & 15) << 2;        // 0..60
      const float* g = xb + (size_t)(c0s + cc) * L + l0 + ll;
      __builtin_amdgcn_global_load_async_to_lds_b128(
          AS1P4(g), AS3P4(&xstage[cc][ll]), 0, 0);
    }
    WAIT_ASYNC();
    __syncthreads();
#pragma unroll
    for (int i = 0; i < 4; ++i) {
      int lin = tid + i * 256;
      int cc  = lin >> 4;
      int ll  = (lin & 15) << 2;
      float4 v = *(const float4*)&xstage[cc][ll];
      xs[ll + 0][c0s + cc] = (__bf16)v.x;
      xs[ll + 1][c0s + cc] = (__bf16)v.y;
      xs[ll + 2][c0s + cc] = (__bf16)v.z;
      xs[ll + 3][c0s + cc] = (__bf16)v.w;
    }
    __syncthreads();
  }
#else
#pragma unroll
  for (int i = 0; i < 16; ++i) {
    int lin = tid + i * 256;            // 0..4095 float4 slots
    int c   = lin >> 4;                 // 0..255
    int l   = (lin & 15) << 2;          // 0..60
    float4 v = *(const float4*)(xb + (size_t)c * L + l0 + l);
    xs[l + 0][c] = (__bf16)v.x;
    xs[l + 1][c] = (__bf16)v.y;
    xs[l + 2][c] = (__bf16)v.z;
    xs[l + 3][c] = (__bf16)v.w;
  }
  __syncthreads();
#endif

  // GEMM1 tiling: wave -> (f-tile mt1 = wave>>1, l-col pair npair = wave&1)
  const int mt1   = wave >> 1;
  const int npair = wave & 1;

  // GEMM2 tiling: wave owns C rows [wave*32, wave*32+32), all 4 l-tiles.
  v8f yacc[2][4];
#pragma unroll
  for (int i = 0; i < 2; ++i)
#pragma unroll
    for (int j = 0; j < 4; ++j) yacc[i][j] = (v8f){};

#pragma unroll 1
  for (int f0 = 0; f0 < F; f0 += FC) {
    const __bf16* aptr = w1e + (size_t)(f0 + mt1 * 16 + lmod) * C;

    // pull next f-chunk's weight fragments toward the WGP early
    if (f0 + FC < F) {
      __builtin_prefetch(aptr + (size_t)FC * C, 0, 3);
      __builtin_prefetch(aptr + (size_t)FC * C + 128, 0, 3);
      const __bf16* wp = w2e + (size_t)(wave * 32 + lmod) * F + f0 + FC;
      __builtin_prefetch(wp, 0, 3);
      __builtin_prefetch(wp + (size_t)16 * F, 0, 3);
    }

    // ======== GEMM1: h[f0+mt1*16 .. +16, npair*32 .. +32] ========
    v8f h0 = (v8f){}, h1 = (v8f){};
#pragma unroll
    for (int c0 = 0; c0 < C; c0 += 32) {
      // A fragment (W1): 16 contiguous K per lane
      v8bf alo = *(const v8bf*)(aptr + c0 + lhalf * 16);
      v8bf ahi = *(const v8bf*)(aptr + c0 + lhalf * 16 + 8);
      v16bf afrag = cat8(alo, ahi);
      // B fragments (x from LDS): 16 contiguous K per lane
      const __bf16* bp0 = &xs[npair * 32 + lmod][c0 + lhalf * 16];
      const __bf16* bp1 = &xs[npair * 32 + 16 + lmod][c0 + lhalf * 16];
      v16bf bf0 = cat8(*(const v8bf*)bp0, *(const v8bf*)(bp0 + 8));
      v16bf bf1 = cat8(*(const v8bf*)bp1, *(const v8bf*)(bp1 + 8));
      h0 = wmma_bf16(afrag, bf0, h0);
      h1 = wmma_bf16(afrag, bf1, h1);
    }

    // bias + silu + pack bf16
    const int mbase = mt1 * 16 + lhalf * 8;   // relative f of acc row r=0
    unsigned int pk0[4], pk1[4];
#pragma unroll
    for (int r = 0; r < 8; ++r) {
      float fb = b1s[f0 + mbase + r];
      float v0 = h0[r] + fb;
      float v1 = h1[r] + fb;
      v0 = v0 / (1.0f + __expf(-v0));
      v1 = v1 / (1.0f + __expf(-v1));
      unsigned int u0 = __builtin_bit_cast(unsigned short, (__bf16)v0);
      unsigned int u1 = __builtin_bit_cast(unsigned short, (__bf16)v1);
      if (r & 1) { pk0[r >> 1] |= u0 << 16; pk1[r >> 1] |= u1 << 16; }
      else       { pk0[r >> 1]  = u0;       pk1[r >> 1]  = u1; }
    }
    const int col0 = npair * 32 + lmod;
    {
      uint4 s0; s0.x = pk0[0]; s0.y = pk0[1]; s0.z = pk0[2]; s0.w = pk0[3];
      uint4 s1; s1.x = pk1[0]; s1.y = pk1[1]; s1.z = pk1[2]; s1.w = pk1[3];
      *(uint4*)(&hs[col0][mbase])      = s0;
      *(uint4*)(&hs[col0 + 16][mbase]) = s1;
    }
    __syncthreads();   // h chunk visible to all waves

    // ======== GEMM2: yacc += W2[:, f0..f0+FC] @ h ========
#pragma unroll
    for (int kk = 0; kk < FC; kk += 32) {
      const __bf16* p0 = w2e + (size_t)(wave * 32 + lmod) * F + f0 + kk + lhalf * 16;
      const __bf16* p1 = p0 + (size_t)16 * F;
      v16bf a0 = cat8(*(const v8bf*)p0, *(const v8bf*)(p0 + 8));
      v16bf a1 = cat8(*(const v8bf*)p1, *(const v8bf*)(p1 + 8));
      v16bf bfr[4];
#pragma unroll
      for (int nt = 0; nt < 4; ++nt) {
        const __bf16* hp = &hs[nt * 16 + lmod][kk + lhalf * 16];
        bfr[nt] = cat8(*(const v8bf*)hp, *(const v8bf*)(hp + 8));
      }
#pragma unroll
      for (int nt = 0; nt < 4; ++nt) {
        yacc[0][nt] = wmma_bf16(a0, bfr[nt], yacc[0][nt]);
        yacc[1][nt] = wmma_bf16(a1, bfr[nt], yacc[1][nt]);
      }
    }
    __syncthreads();   // done reading hs before next chunk overwrites it
  }

  // ======== epilogue: y = yacc + b2 ========
#pragma unroll
  for (int mt = 0; mt < 2; ++mt) {
    const int mrow = wave * 32 + mt * 16 + lhalf * 8;
#pragma unroll
    for (int nt = 0; nt < 4; ++nt) {
      const int lcol = l0 + nt * 16 + lmod;
#pragma unroll
      for (int r = 0; r < 8; ++r) {
        yb[(size_t)(mrow + r) * L + lcol] = yacc[mt][nt][r] + b2s[mrow + r];
      }
    }
  }
}

extern "C" void kernel_launch(void* const* d_in, const int* in_sizes, int n_in,
                              void* d_out, int out_size, void* d_ws, size_t ws_size,
                              hipStream_t stream) {
  const float* x  = (const float*)d_in[0];
  const float* W1 = (const float*)d_in[1];
  const float* b1 = (const float*)d_in[2];
  const float* W2 = (const float*)d_in[3];
  const float* b2 = (const float*)d_in[4];
  float* y = (float*)d_out;

  __bf16* w1b = (__bf16*)d_ws;                 // E*F*C bf16
  __bf16* w2b = w1b + (size_t)E * F * C;       // E*C*F bf16

  const int n1 = E * F * C;                    // 4,194,304
  const int n2 = E * C * F;                    // 4,194,304
  cvt_f32_to_bf16<<<n1 / 1024, 256, 0, stream>>>(W1, w1b, n1);
  cvt_f32_to_bf16<<<n2 / 1024, 256, 0, stream>>>(W2, w2b, n2);

  dim3 grid(BSZ * E * (L / LT));               // 2048 workgroups
  moe_fused<<<grid, 256, 0, stream>>>(x, w1b, w2b, b1, b2, y);
}
